// AggregateAttention_678604833095
// MI455X (gfx1250) — compile-verified, matrix operand order checked by reading, not run on hardware
//
#include <hip/hip_runtime.h>

// Problem dims
#define B_  64
#define A_  6
#define R_  29
#define D_  512
#define N_  1000
#define NP_ 1024   // N padded to multiple of 64 in workspace

typedef __attribute__((ext_vector_type(16))) __bf16 v16bf;
typedef __attribute__((ext_vector_type(8)))  float  v8f;

static __device__ __forceinline__ v8f wmma_bf16(v16bf a, v16bf b, v8f c) {
  // (neg_a, A, neg_b, B, c_mod, C, reuse_a, reuse_b)
  return __builtin_amdgcn_wmma_f32_16x16x32_bf16(false, a, false, b, (short)0, c,
                                                 false, false);
}

// A-style fragment: 16 (M) x 32 (K) bf16 from row-major f32, row stride ld.
// Lane layout per ISA 7.12.2: lanes 0-15 row=lane K=0..7,16..23; lanes 16-31 K=8..15,24..31.
static __device__ __forceinline__ v16bf frag_rows_f32(const float* __restrict__ base,
                                                      long ld, int lane, int rowsValid) {
  int row = lane & 15;
  int k0  = (lane & 16) ? 8 : 0;
  v16bf f;
  if (row < rowsValid) {
    const float* p = base + (long)row * ld + k0;
#pragma unroll
    for (int i = 0; i < 8; ++i) f[i]     = static_cast<__bf16>(p[i]);
#pragma unroll
    for (int i = 0; i < 8; ++i) f[8 + i] = static_cast<__bf16>(p[16 + i]);
  } else {
#pragma unroll
    for (int i = 0; i < 16; ++i) f[i] = static_cast<__bf16>(0.0f);
  }
  return f;
}

// Same lane layout, bf16 source (global workspace or LDS).
static __device__ __forceinline__ v16bf frag_rows_bf16(const __bf16* __restrict__ base,
                                                       long ld, int lane) {
  int row = lane & 15;
  int k0  = (lane & 16) ? 8 : 0;
  const __bf16* p = base + (long)row * ld + k0;
  v16bf f;
#pragma unroll
  for (int i = 0; i < 8; ++i) f[i]     = p[i];
#pragma unroll
  for (int i = 0; i < 8; ++i) f[8 + i] = p[16 + i];
  return f;
}

// B-style fragment: 32 (K) x 16 (N), column per lane, from row-major f32 [K][ld].
static __device__ __forceinline__ v16bf frag_cols_f32(const float* __restrict__ base,
                                                      long ld, int lane, int kValid) {
  int col = lane & 15;
  int k0  = (lane & 16) ? 8 : 0;
  v16bf f;
#pragma unroll
  for (int i = 0; i < 8; ++i) {
    int k = k0 + i;
    f[i] = (k < kValid) ? static_cast<__bf16>(base[(long)k * ld + col])
                        : static_cast<__bf16>(0.0f);
  }
#pragma unroll
  for (int i = 0; i < 8; ++i) {
    int k = k0 + 16 + i;
    f[8 + i] = (k < kValid) ? static_cast<__bf16>(base[(long)k * ld + col])
                            : static_cast<__bf16>(0.0f);
  }
  return f;
}

// ---------------------------------------------------------------------------
// Kernel 1: y_wy[r,n,e] = pool[r,n,:] @ wy[r,:,e] + wy_bias[r,e]  -> bf16 ws
// grid (etile=8, ntile=16, r=29), 256 threads; block computes 64x64 output.
// ---------------------------------------------------------------------------
__global__ __launch_bounds__(256) void proj_pool_kernel(const float* __restrict__ pool,
                                                        const float* __restrict__ wy,
                                                        const float* __restrict__ wyb,
                                                        __bf16* __restrict__ ybf) {
  const int r = blockIdx.z, nt = blockIdx.y, et = blockIdx.x;
  const int wave = threadIdx.x >> 5, lane = threadIdx.x & 31;
  const int n0 = nt * 64 + (wave & 3) * 16;
  const int e0 = et * 64 + (wave >> 2) * 32;
  const float* Ab = pool + ((long)r * N_ + n0) * D_;
  int rowsValid = N_ - n0;
  if (rowsValid > 16) rowsValid = 16;
  if (rowsValid < 0)  rowsValid = 0;
  const float* Wb = wy + (long)r * D_ * D_ + e0;
  v8f acc0 = {}; v8f acc1 = {};
  for (int k = 0; k < D_; k += 32) {
    if (k + 32 < D_) __builtin_prefetch(Ab + k + 32, 0, 1);
    v16bf a  = frag_rows_f32(Ab + k, D_, lane, rowsValid);
    v16bf b0 = frag_cols_f32(Wb + (long)k * D_,      D_, lane, 32);
    v16bf b1 = frag_cols_f32(Wb + (long)k * D_ + 16, D_, lane, 32);
    acc0 = wmma_bf16(a, b0, acc0);
    acc1 = wmma_bf16(a, b1, acc1);
  }
  const int col = lane & 15, rb = (lane & 16) ? 8 : 0;
  const float bias0 = wyb[(long)r * D_ + e0 + col];
  const float bias1 = wyb[(long)r * D_ + e0 + 16 + col];
#pragma unroll
  for (int v = 0; v < 8; ++v) {
    int n = n0 + rb + v;
    bool ok = (n < N_);
    __bf16* o = ybf + ((long)r * NP_ + n) * D_ + e0 + col;
    o[0]  = static_cast<__bf16>(ok ? acc0[v] + bias0 : 0.0f);   // zero the pad rows
    o[16] = static_cast<__bf16>(ok ? acc1[v] + bias1 : 0.0f);
  }
}

// ---------------------------------------------------------------------------
// Kernel 2: x_wx[a,r,b,e] = top[b,r,:] @ wx[a,r,:,e] + wx_bias[a,r,e] -> bf16 ws
// grid (etile=8, a=6, r=29), 256 threads; block computes 64x64 output.
// ---------------------------------------------------------------------------
__global__ __launch_bounds__(256) void proj_x_kernel(const float* __restrict__ top,
                                                     const float* __restrict__ wx,
                                                     const float* __restrict__ wxb,
                                                     __bf16* __restrict__ xbf) {
  const int r = blockIdx.z, aa = blockIdx.y, et = blockIdx.x;
  const int wave = threadIdx.x >> 5, lane = threadIdx.x & 31;
  const int b0 = (wave & 3) * 16;
  const int e0 = et * 64 + (wave >> 2) * 32;
  const float* Ab = top + (long)b0 * (R_ * D_) + (long)r * D_;   // row stride R*D
  const float* Wb = wx + ((long)aa * R_ + r) * D_ * D_ + e0;
  v8f acc0 = {}; v8f acc1 = {};
  for (int k = 0; k < D_; k += 32) {
    if (k + 32 < D_) __builtin_prefetch(Wb + (long)(k + 32) * D_, 0, 1);
    v16bf a  = frag_rows_f32(Ab + k, (long)R_ * D_, lane, 16);
    v16bf b0f = frag_cols_f32(Wb + (long)k * D_,      D_, lane, 32);
    v16bf b1f = frag_cols_f32(Wb + (long)k * D_ + 16, D_, lane, 32);
    acc0 = wmma_bf16(a, b0f, acc0);
    acc1 = wmma_bf16(a, b1f, acc1);
  }
  const int col = lane & 15, rb = (lane & 16) ? 8 : 0;
  const float bias0 = wxb[((long)aa * R_ + r) * D_ + e0 + col];
  const float bias1 = wxb[((long)aa * R_ + r) * D_ + e0 + 16 + col];
#pragma unroll
  for (int v = 0; v < 8; ++v) {
    int b = b0 + rb + v;
    __bf16* o = xbf + (((long)aa * R_ + r) * B_ + b) * D_ + e0 + col;
    o[0]  = static_cast<__bf16>(acc0[v] + bias0);
    o[16] = static_cast<__bf16>(acc1[v] + bias1);
  }
}

// ---------------------------------------------------------------------------
// Kernel 3: flash-style fused attention per (a,r):
//   S = X(64xD) @ Y^T / sqrt(D); P = online-softmax(S); O = P @ pool_r
// 512 threads = 16 waves. Each wave owns 32 output columns (O: 4x2 f32 tiles)
// and computes one 16x16 score tile per 64-wide n-chunk.
// ---------------------------------------------------------------------------
__global__ __launch_bounds__(512) void attn_kernel(const float* __restrict__ pool,
                                                   const __bf16* __restrict__ xbf,
                                                   const __bf16* __restrict__ ybf,
                                                   float* __restrict__ out) {
  const int aa = blockIdx.x, r = blockIdx.y;
  const int wave = threadIdx.x >> 5, lane = threadIdx.x & 31;
  const int tid = threadIdx.x;

  __shared__ float  S[64][72];      // padded stride vs 64 banks
  __shared__ __bf16 P[64][64];
  __shared__ float  mrow[64], lrow[64], arow[64];

  const __bf16* X = xbf + ((long)aa * R_ + r) * B_ * D_;
  const __bf16* Y = ybf + (long)r * NP_ * D_;
  const float*  V = pool + (long)r * N_ * D_;

  const int ebase = wave * 32;            // this wave's output columns
  const int smt = wave & 3, snt = wave >> 2;
  const int col = lane & 15, rb = (lane & 16) ? 8 : 0;

  v8f zero = {};
  v8f O[4][2];
#pragma unroll
  for (int mi = 0; mi < 4; ++mi)
#pragma unroll
    for (int ei = 0; ei < 2; ++ei) O[mi][ei] = zero;

  if (tid < 64) { mrow[tid] = -3.0e38f; lrow[tid] = 0.0f; }
  __syncthreads();

  for (int chunk = 0; chunk < 16; ++chunk) {
    const int n0 = chunk * 64;

    // ---- score tile: rows smt*16.., pool-cols n0+snt*16.. (K = D) ----
    v8f s = zero;
    const __bf16* Arow = X + (long)(smt * 16) * D_;
    const __bf16* Brow = Y + (long)(n0 + snt * 16) * D_;   // rows of Y == cols of Y^T
    for (int k = 0; k < D_; k += 32) {
      if (k + 32 < D_) __builtin_prefetch(Brow + k + 32, 0, 1);
      v16bf af = frag_rows_bf16(Arow + k, D_, lane);
      v16bf bf = frag_rows_bf16(Brow + k, D_, lane);
      s = wmma_bf16(af, bf, s);
    }
    const float scale = 0.044194173824159216f;  // 1/sqrt(512)
#pragma unroll
    for (int v = 0; v < 8; ++v)
      S[smt * 16 + rb + v][snt * 16 + col] = s[v] * scale;
    __syncthreads();

    // ---- online softmax update, one thread per output row ----
    if (tid < 64) {
      const int row = tid;
      float mold = mrow[row];
      float mx = mold;
      for (int c = 0; c < 64; ++c) {
        float sv = (n0 + c < N_) ? S[row][c] : -3.0e38f;
        S[row][c] = sv;
        mx = fmaxf(mx, sv);
      }
      float alpha = __expf(mold - mx);
      float sum = 0.0f;
      for (int c = 0; c < 64; ++c) {
        float p = __expf(S[row][c] - mx);
        P[row][c] = static_cast<__bf16>(p);
        sum += p;
      }
      mrow[row] = mx;
      lrow[row] = alpha * lrow[row] + sum;
      arow[row] = alpha;
    }
    __syncthreads();

    // ---- rescale accumulators, then O += P @ V_chunk ----
#pragma unroll
    for (int mi = 0; mi < 4; ++mi) {
      float af8[8];
#pragma unroll
      for (int v = 0; v < 8; ++v) af8[v] = arow[mi * 16 + rb + v];
#pragma unroll
      for (int ei = 0; ei < 2; ++ei)
#pragma unroll
        for (int v = 0; v < 8; ++v) O[mi][ei][v] *= af8[v];
    }
#pragma unroll
    for (int ks = 0; ks < 2; ++ks) {
      const int kn = ks * 32;
      const int nValid = N_ - (n0 + kn);   // real pool rows in this K-slab
#pragma unroll
      for (int mi = 0; mi < 4; ++mi) {
        v16bf pf = frag_rows_bf16(&P[mi * 16][kn], 64, lane);
#pragma unroll
        for (int ei = 0; ei < 2; ++ei) {
          v16bf vf = frag_cols_f32(V + (long)(n0 + kn) * D_ + ebase + ei * 16,
                                   D_, lane, nValid);
          O[mi][ei] = wmma_bf16(pf, vf, O[mi][ei]);
        }
      }
    }
    __syncthreads();
  }

  // ---- normalize and store, output layout (b, r, a, d) ----
#pragma unroll
  for (int mi = 0; mi < 4; ++mi) {
    float lf[8];
#pragma unroll
    for (int v = 0; v < 8; ++v) lf[v] = lrow[mi * 16 + rb + v];
#pragma unroll
    for (int ei = 0; ei < 2; ++ei)
#pragma unroll
      for (int v = 0; v < 8; ++v) {
        int b = mi * 16 + rb + v;
        int e = ebase + ei * 16 + col;
        out[(((long)b * R_ + r) * A_ + aa) * D_ + e] = O[mi][ei][v] / lf[v];
      }
  }
}

// ---------------------------------------------------------------------------
extern "C" void kernel_launch(void* const* d_in, const int* in_sizes, int n_in,
                              void* d_out, int out_size, void* d_ws, size_t ws_size,
                              hipStream_t stream) {
  const float* top  = (const float*)d_in[0];   // (B,R,D)
  const float* pool = (const float*)d_in[1];   // (R,N,D)
  const float* wx   = (const float*)d_in[2];   // (A,R,D,D)
  const float* wxb  = (const float*)d_in[3];   // (A,R,1,D)
  const float* wy   = (const float*)d_in[4];   // (R,D,D)
  const float* wyb  = (const float*)d_in[5];   // (R,1,D)
  float* out = (float*)d_out;                  // (B,R,A,D) f32

  // Workspace: y_wy bf16 [R][NP][D] then x_wx bf16 [A][R][B][D]
  __bf16* ybf = (__bf16*)d_ws;
  __bf16* xbf = (__bf16*)((char*)d_ws + (size_t)R_ * NP_ * D_ * sizeof(__bf16));

  proj_pool_kernel<<<dim3(D_ / 64, NP_ / 64, R_), 256, 0, stream>>>(pool, wy, wyb, ybf);
  proj_x_kernel  <<<dim3(D_ / 64, A_, R_),       256, 0, stream>>>(top, wx, wxb, xbf);
  attn_kernel    <<<dim3(A_, R_),                512, 0, stream>>>(pool, xbf, ybf, out);

  (void)in_sizes; (void)n_in; (void)out_size; (void)ws_size;
}
